// GNNModel_30064771072295
// MI455X (gfx1250) — matrix operand order repaired and checked by reading, not compiled, not run on hardware
//
#include <hip/hip_runtime.h>
#include <hip/hip_bf16.h>

// GCN forward for MI455X (gfx1250, wave32, WMMA).
//
// Roofline: dominant phase is edge aggregation (~102M f32 atomic-adds into a
// 25.6MB buffer that lives in the 192MB L2); HBM floor ~150MB ~= 7us @23.3TB/s.
// GEMMs (2.5 GFLOP) use v_wmma_f32_16x16x32_bf16 with LDS-staged, bf16-
// converted tiles (A and B fragments are aligned ds_load_b128 pairs).

typedef __attribute__((ext_vector_type(16))) __bf16 v16bf;
typedef __attribute__((ext_vector_type(8)))  __bf16 v8bf;
typedef __attribute__((ext_vector_type(8)))  float  v8f;

#define IN_DIM  128
#define HID_DIM 64

union AFrag { v16bf v; v8bf h[2]; };

// ---------------- degree / normalization ----------------

__global__ void deg_init(float* __restrict__ deg, int n) {
    int i = blockIdx.x * 256 + threadIdx.x;
    if (i < n) deg[i] = 1.0f;              // self loop
}

__global__ void deg_count(const int* __restrict__ dst, float* __restrict__ deg, int e) {
    int i = blockIdx.x * 256 + threadIdx.x;
    if (i < e) atomicAdd(&deg[dst[i]], 1.0f);
}

__global__ void deg_rsqrt(float* __restrict__ deg, int n) {
    int i = blockIdx.x * 256 + threadIdx.x;
    if (i < n) deg[i] = rsqrtf(deg[i]);    // deg > 0 always (self loop)
}

// ---------------- WMMA GEMM:  H[N,64] = act(X)[N,K] @ W[K,64] ----------------
// 256 threads = 8 waves; wave w -> colTile = w&3 (4x16 = 64 cols),
// rowTile = w>>2 (2x16 = 32 rows/block). K in {128,64} fully unrolled by 32.
// X tile (32xK) and W (transposed, K+8 pad) staged in LDS as bf16; ReLU of the
// previous layer fused into the staging convert (RELU template flag).

template<int K, bool RELU>
__global__ __launch_bounds__(256) void gemm_wmma_bf16(
    const float* __restrict__ X, const float* __restrict__ W,
    float* __restrict__ H)
{
    __shared__ __bf16 sWt[HID_DIM * (K + 8)];  // [n][k], padded vs bank conflicts
    __shared__ __bf16 sX [32 * K];             // [r][k], 32-row tile

    const int tid = threadIdx.x;
    const size_t rowBlk = (size_t)blockIdx.x * 32;

    // stage W transposed (one-time, bf16)
    for (int i = tid; i < K * HID_DIM; i += 256) {
        int k = i >> 6, n = i & 63;
        sWt[n * (K + 8) + k] = (__bf16)W[i];
    }
    // stage X tile (contiguous 32*K floats), fusing previous layer's ReLU
    const float* xt = X + rowBlk * K;
    for (int i = tid; i < 32 * K; i += 256) {
        float v = xt[i];
        if (RELU) v = fmaxf(v, 0.0f);
        sX[i] = (__bf16)v;
    }
    __syncthreads();

    const int lane = tid & 31;
    const int wave = tid >> 5;
    const int hi   = lane >> 4;                       // half-wave id
    const int n    = ((wave & 3) << 4) + (lane & 15); // output column
    const int rloc = ((wave >> 2) << 4) + (lane & 15);// A row within tile

    v8f c = {0.f, 0.f, 0.f, 0.f, 0.f, 0.f, 0.f, 0.f};

    #pragma unroll
    for (int k0 = 0; k0 < K; k0 += 32) {
        // A 16x32 bf16 (ISA 7.12.2): K = 16*(j>>3) + 8*hi + (j&7)
        //   -> two 16B-aligned 8-element runs => ds_load_b128 x2
        AFrag a;
        const __bf16* ap = sX + rloc * K + k0 + (hi << 3);
        a.h[0] = *(const v8bf*)(ap);
        a.h[1] = *(const v8bf*)(ap + 16);
        // B 32x16 bf16: lane holds col n; K = 16*(j>>3) + 8*hi + (j&7) (same as A)
        AFrag b;
        const __bf16* bp = sWt + n * (K + 8) + k0 + (hi << 3);
        b.h[0] = *(const v8bf*)(bp);
        b.h[1] = *(const v8bf*)(bp + 16);
        c = __builtin_amdgcn_wmma_f32_16x16x32_bf16(
                /*neg_a=*/false, a.v, /*neg_b=*/false, b.v,
                /*c_mod=*/(short)0, c, /*reuse_a=*/false, /*reuse_b=*/false);
    }

    // C/D layout: VGPR v -> row (v + 8*hi), col n. N%32==0 -> no guards.
    float* hp = H + (rowBlk + ((wave >> 2) << 4) + (hi << 3)) * HID_DIM + n;
    #pragma unroll
    for (int v = 0; v < 8; ++v)
        hp[(size_t)v * HID_DIM] = c[v];
}

// ---------------- aggregation ----------------

// out[node,f] = h[node,f]*dis[node]^2 + b[f]   (full overwrite -> poison-safe)
__global__ void self_bias64(const float* __restrict__ h, const float* __restrict__ dis,
                            const float* __restrict__ b, float* __restrict__ out, int n) {
    int i = blockIdx.x * 256 + threadIdx.x;   // over n*64
    if (i < n * HID_DIM) {
        int node = i >> 6;
        float d = dis[node];
        out[i] = h[i] * d * d + b[i & 63];
    }
}

// out[dst,f] += h[src,f]*dis[src]*dis[dst]; 16 threads/edge, float4 per thread
__global__ void scatter64(const float* __restrict__ h, const float* __restrict__ dis,
                          const int* __restrict__ src, const int* __restrict__ dst,
                          float* __restrict__ out, int e) {
    int i = blockIdx.x * 256 + threadIdx.x;   // e*16 = 25.6M, fits int
    if (i >= e * 16) return;
    int ed = i >> 4, f4 = (i & 15) << 2;
    int s = src[ed], d = dst[ed];
    float norm = dis[s] * dis[d];
    const float4 hv = *(const float4*)(h + (size_t)s * HID_DIM + f4);
    float* op = out + (size_t)d * HID_DIM + f4;
    atomicAdd(op + 0, hv.x * norm);
    atomicAdd(op + 1, hv.y * norm);
    atomicAdd(op + 2, hv.z * norm);
    atomicAdd(op + 3, hv.w * norm);
}

// ---------------- layer 3 (OUT_DIM = 1), ReLU fused ----------------

__global__ void dot64_relu(const float* __restrict__ a, const float* __restrict__ w3,
                           float* __restrict__ h3, int n) {
    int node = blockIdx.x * 256 + threadIdx.x;
    if (node >= n) return;
    const float4* r = (const float4*)(a + (size_t)node * HID_DIM);
    float acc = 0.f;
    #pragma unroll
    for (int q = 0; q < 16; ++q) {
        float4 v = r[q];
        acc += fmaxf(v.x, 0.f) * w3[q * 4 + 0] + fmaxf(v.y, 0.f) * w3[q * 4 + 1]
             + fmaxf(v.z, 0.f) * w3[q * 4 + 2] + fmaxf(v.w, 0.f) * w3[q * 4 + 3];
    }
    h3[node] = acc;
}

__global__ void out_init1(const float* __restrict__ h3, const float* __restrict__ dis,
                          const float* __restrict__ b3, float* __restrict__ out, int n) {
    int i = blockIdx.x * 256 + threadIdx.x;
    if (i < n) { float d = dis[i]; out[i] = h3[i] * d * d + b3[0]; }
}

__global__ void scatter1(const float* __restrict__ h3, const float* __restrict__ dis,
                         const int* __restrict__ src, const int* __restrict__ dst,
                         float* __restrict__ out, int e) {
    int i = blockIdx.x * 256 + threadIdx.x;
    if (i < e) {
        int s = src[i], d = dst[i];
        atomicAdd(&out[d], h3[s] * dis[s] * dis[d]);
    }
}

// ---------------- launcher ----------------

extern "C" void kernel_launch(void* const* d_in, const int* in_sizes, int n_in,
                              void* d_out, int out_size, void* d_ws, size_t ws_size,
                              hipStream_t stream) {
    (void)n_in; (void)out_size; (void)ws_size;
    const float* x   = (const float*)d_in[0];
    const int*   ei  = (const int*)d_in[1];
    const float* W1  = (const float*)d_in[2];
    const float* b1  = (const float*)d_in[3];
    const float* W2  = (const float*)d_in[4];
    const float* b2  = (const float*)d_in[5];
    const float* W3  = (const float*)d_in[6];
    const float* b3  = (const float*)d_in[7];
    float* out = (float*)d_out;

    const int N = in_sizes[0] / IN_DIM;     // 100000 (divisible by 32)
    const int E = in_sizes[1] / 2;          // 1600000
    const int* src = ei;
    const int* dst = ei + E;

    // workspace: dis | A (h pre-agg) | B (agg/activation) | h3  (~52MB)
    float* ws  = (float*)d_ws;
    float* dis = ws;
    float* A   = ws + (((size_t)N + 255) & ~(size_t)255);
    float* B   = A + (size_t)N * HID_DIM;
    float* h3  = B + (size_t)N * HID_DIM;

    auto cdiv = [](long long a, long long b) { return (unsigned)((a + b - 1) / b); };
    const unsigned gN   = cdiv(N, 256);
    const unsigned gE   = cdiv(E, 256);
    const unsigned gNF  = cdiv((long long)N * HID_DIM, 256);
    const unsigned gE4  = cdiv((long long)E * 16, 256);
    const unsigned gMM  = cdiv(N, 32);

    // normalization
    deg_init <<<gN, 256, 0, stream>>>(dis, N);
    deg_count<<<gE, 256, 0, stream>>>(dst, dis, E);
    deg_rsqrt<<<gN, 256, 0, stream>>>(dis, N);

    // layer 1: h = x@W1 ; self+bias ; edge scatter   (ReLU deferred into L2 GEMM)
    gemm_wmma_bf16<IN_DIM,  false><<<gMM, 256, 0, stream>>>(x, W1, A);
    self_bias64<<<gNF, 256, 0, stream>>>(A, dis, b1, B, N);
    scatter64  <<<gE4, 256, 0, stream>>>(A, dis, src, dst, B, E);

    // layer 2: h = relu(a1)@W2 ; self+bias ; edge scatter (ReLU deferred into dot)
    gemm_wmma_bf16<HID_DIM, true ><<<gMM, 256, 0, stream>>>(B, W2, A);
    self_bias64<<<gNF, 256, 0, stream>>>(A, dis, b2, B, N);
    scatter64  <<<gE4, 256, 0, stream>>>(A, dis, src, dst, B, E);

    // layer 3 (scalar output): h3 = relu(a2)@W3 ; self+bias ; edge scatter
    dot64_relu<<<gN, 256, 0, stream>>>(B, W3, h3, N);
    out_init1 <<<gN, 256, 0, stream>>>(h3, dis, b3, out, N);
    scatter1  <<<gE, 256, 0, stream>>>(h3, dis, src, dst, out, E);
}
